// LigerFusedLinearDPOLoss_23991687315566
// MI455X (gfx1250) — matrix operand order repaired and checked by reading, not compiled, not run on hardware
//
#include <hip/hip_runtime.h>
#include <hip/hip_bf16.h>
#include <math.h>

#define HIDDEN 4096
#define VOCAB  32000
#define TWO_B  8
#define SEQ    512
#define MTOT   (TWO_B * SEQ)   // 4096 token rows
#define MT     256             // M tile per workgroup
#define NT     256             // N tile per workgroup
#define KT     32              // K step (bf16 WMMA K)
#define NK     (HIDDEN / KT)   // 128 K steps
#define LDSTR  20              // LDS row stride in dwords (16 used + 4 pad; 16B aligned, conflict-free)
#define IGNORE_INDEX (-100)
#define BETA   0.1f

typedef __bf16 bf16_t;
typedef __attribute__((ext_vector_type(16))) bf16_t v16bf;
typedef __attribute__((ext_vector_type(8)))  float  v8f;

// gfx1250 async global->LDS copy path (ASYNCcnt), with safe fallback
#if defined(__gfx1250__) && __has_builtin(__builtin_amdgcn_global_load_async_to_lds_b128)
#define USE_ASYNC_LDS 1
typedef int v4i_t __attribute__((ext_vector_type(4)));
typedef __attribute__((address_space(1))) v4i_t as1_v4i;   // global (prints as __device__)
typedef __attribute__((address_space(3))) v4i_t as3_v4i;   // LDS
__device__ __forceinline__ void wait_asynccnt0() {
#if __has_builtin(__builtin_amdgcn_s_wait_asynccnt)
    __builtin_amdgcn_s_wait_asynccnt(0);
#else
    asm volatile("s_wait_asynccnt 0x0" ::: "memory");
#endif
}
#else
#define USE_ASYNC_LDS 0
#endif

union Frag {
    unsigned u[8];
    uint4    q[2];
    v16bf    v;
};

// round-to-nearest-even f32 -> bf16, packed pair into one dword
__device__ __forceinline__ unsigned pack_bf16(float a, float b) {
    unsigned ua = __float_as_uint(a), ub = __float_as_uint(b);
    ua += 0x7FFFu + ((ua >> 16) & 1u);
    ub += 0x7FFFu + ((ub >> 16) & 1u);
    return (ua >> 16) | (ub & 0xFFFF0000u);
}

// ---------------------------------------------------------------------------
// One-shot f32 -> packed bf16 stream conversion (8 elements / thread)
// ---------------------------------------------------------------------------
__global__ __launch_bounds__(256)
void cvt_bf16_kernel(const float* __restrict__ src,
                     unsigned short* __restrict__ dst, int n8) {
    int i = blockIdx.x * blockDim.x + threadIdx.x;
    if (i < n8) {
        const float4* p = reinterpret_cast<const float4*>(src) + (size_t)i * 2;
        float4 a = p[0], b = p[1];
        reinterpret_cast<uint4*>(dst)[i] =
            make_uint4(pack_bf16(a.x, a.y), pack_bf16(a.z, a.w),
                       pack_bf16(b.x, b.y), pack_bf16(b.z, b.w));
    }
}

// ---------------------------------------------------------------------------
// zero per-token accumulators (ws poisoned; must be replay-safe)
// ---------------------------------------------------------------------------
__global__ void dpo_zero_kernel(float* __restrict__ sumexp,
                                float* __restrict__ tlogit, int n) {
    int i = blockIdx.x * blockDim.x + threadIdx.x;
    if (i < n) { sumexp[i] = 0.f; tlogit[i] = 0.f; }
}

// ---------------------------------------------------------------------------
// bf16 WMMA GEMM (f32 acc) + bias + online sum-exp + target logit pick
//   grid = (MTOT/MT=16, VOCAB/NT=125), block = 512 (16 wave32, 4x4 wave grid)
//   Each wave: 64(M) x 64(N) = 16 accumulator tiles -> 128 C VGPRs.
//   K loop kept at unroll 1 with pointer-swapped double buffers so the
//   accumulator PHIs coalesce (no v_mov rotation, no spills).
// ---------------------------------------------------------------------------
__global__ __launch_bounds__(512)
void dpo_gemm_kernel(const unsigned short* __restrict__ Wb,  // bf16 [VOCAB][HIDDEN]
                     const unsigned short* __restrict__ Xb,  // bf16 [MTOT][HIDDEN]
                     const int*   __restrict__ tgt,          // [MTOT]
                     const float* __restrict__ bias,         // [VOCAB]
                     float* __restrict__ sumexp,             // [MTOT]
                     float* __restrict__ tlogit)             // [MTOT]
{
    __shared__ unsigned lds_a[2][MT * LDSTR];   // 2 x 20 KB
    __shared__ unsigned lds_b[2][NT * LDSTR];   // 2 x 20 KB

    const int m0   = blockIdx.x * MT;
    const int n0   = blockIdx.y * NT;
    const int tid  = threadIdx.x;               // 0..511
    const int lane = tid & 31;
    const int w    = tid >> 5;                  // 0..15
    const int wm   = (w & 3) * 64;              // 4 wave strips of 64 rows (M)
    const int wn   = (w >> 2) * 64;             // 4 wave strips of 64 cols (N)

    // per-lane WMMA layout helpers (ISA 7.12.2)
    const int nlan    = lane & 15;
    const int halfsel = lane >> 4;
    const int kb_a    = halfsel ? 8 : 0;        // A: kk offset of the 8-dword run
    const int kb_b    = halfsel ? 8 : 0;        // B: kk offset of the 8-dword run

    // staging: one 64B row per thread; tid<256 -> A rows, tid>=256 -> B rows
    const bool isA = (tid < 256);
    const int  row = isA ? tid : (tid - 256);
    const unsigned short* gRow =
        isA ? (Xb + (size_t)(m0 + row) * HIDDEN)
            : (Wb + (size_t)(n0 + row) * HIDDEN);
    unsigned* lw0 = isA ? &lds_a[0][row * LDSTR] : &lds_b[0][row * LDSTR];
    unsigned* lw1 = isA ? &lds_a[1][row * LDSTR] : &lds_b[1][row * LDSTR];

    // per-wave fragment read bases for both buffers
    const unsigned* ar  = &lds_a[0][(wm + nlan) * LDSTR];
    const unsigned* arN = &lds_a[1][(wm + nlan) * LDSTR];
    const unsigned* br  = &lds_b[0][(wn + nlan) * LDSTR];
    const unsigned* brN = &lds_b[1][(wn + nlan) * LDSTR];

    v8f c[4][4];
    for (int i = 0; i < 4; ++i)
        for (int j = 0; j < 4; ++j)
            c[i][j] = (v8f)(0.0f);

#if USE_ASYNC_LDS
    // stage one 64B row into LDS: IOFFSET applies to BOTH global and LDS addr (GV mode)
    #define STAGE_ROW(gptr, lptr)                                                     \
        do {                                                                          \
            as1_v4i* _g = (as1_v4i*)(gptr);                                           \
            as3_v4i* _l = (as3_v4i*)(lptr);                                           \
            __builtin_amdgcn_global_load_async_to_lds_b128(_g, _l, 0,  0);            \
            __builtin_amdgcn_global_load_async_to_lds_b128(_g, _l, 16, 0);            \
            __builtin_amdgcn_global_load_async_to_lds_b128(_g, _l, 32, 0);            \
            __builtin_amdgcn_global_load_async_to_lds_b128(_g, _l, 48, 0);            \
        } while (0)

    // ---- prologue: stage K-step 0 into buffer 0 ----
    STAGE_ROW(gRow, lw0);
    wait_asynccnt0();
    __syncthreads();
#else
    {
        const uint4* pg = reinterpret_cast<const uint4*>(gRow);
        uint4 r0 = pg[0], r1 = pg[1], r2 = pg[2], r3 = pg[3];
        *reinterpret_cast<uint4*>(lw0 + 0)  = r0;
        *reinterpret_cast<uint4*>(lw0 + 4)  = r1;
        *reinterpret_cast<uint4*>(lw0 + 8)  = r2;
        *reinterpret_cast<uint4*>(lw0 + 12) = r3;
    }
    __syncthreads();
#endif

    unsigned* lw  = lw1;   // write target for the NEXT K step
    unsigned* lwN = lw0;

#pragma unroll 1
    for (int ks = 0; ks < NK; ++ks) {
        const bool pre = (ks + 1 < NK);

#if USE_ASYNC_LDS
        // ---- issue next tile's async copies first: overlap with the WMMAs ----
        if (pre) STAGE_ROW(gRow + (ks + 1) * KT, lw);
#else
        uint4 r0, r1, r2, r3;
        if (pre) {
            const uint4* pg = reinterpret_cast<const uint4*>(gRow + (ks + 1) * KT);
            r0 = pg[0]; r1 = pg[1]; r2 = pg[2]; r3 = pg[3];
        }
#endif

        // ---- B fragments for all 4 column tiles (two ds_load_b128 each) ----
        Frag bfr[4];
        for (int nt = 0; nt < 4; ++nt) {
            const unsigned* base = br + nt * 16 * LDSTR;
            bfr[nt].q[0] = *reinterpret_cast<const uint4*>(base + kb_b);
            bfr[nt].q[1] = *reinterpret_cast<const uint4*>(base + kb_b + 4);
        }

        // ---- A fragment per row tile, 4 WMMAs each (16 WMMAs / wave / K-step) ----
        for (int mt = 0; mt < 4; ++mt) {
            Frag a;
            const unsigned* base = ar + mt * 16 * LDSTR;
            a.q[0] = *reinterpret_cast<const uint4*>(base + kb_a);
            a.q[1] = *reinterpret_cast<const uint4*>(base + kb_a + 4);
            for (int nt = 0; nt < 4; ++nt)
                c[mt][nt] = __builtin_amdgcn_wmma_f32_16x16x32_bf16(
                    false, a.v, false, bfr[nt].v,
                    (short)0, c[mt][nt], false, false);
        }

#if USE_ASYNC_LDS
        if (pre) wait_asynccnt0();   // async copies landed in LDS before barrier
#else
        if (pre) {
            *reinterpret_cast<uint4*>(lw + 0)  = r0;
            *reinterpret_cast<uint4*>(lw + 4)  = r1;
            *reinterpret_cast<uint4*>(lw + 8)  = r2;
            *reinterpret_cast<uint4*>(lw + 12) = r3;
        }
#endif
        __syncthreads();

        // ---- swap double buffers (loop-carried pointers, unroll 1) ----
        const unsigned* t;
        t = ar; ar = arN; arN = t;
        t = br; br = brN; brN = t;
        unsigned* t2 = lw; lw = lwN; lwN = t2;
    }

    // ---- epilogue: nt-outer so each column strip's C tiles die progressively ----
    // per-row targets (same address across the 16 lanes of a half -> broadcast load)
    int tgrow[4][8];
    for (int mt = 0; mt < 4; ++mt)
        for (int r = 0; r < 8; ++r)
            tgrow[mt][r] = tgt[m0 + wm + mt * 16 + r + halfsel * 8];

    float sacc[4][8];                       // partial sum of exp over this lane's columns
    for (int mt = 0; mt < 4; ++mt)
        for (int r = 0; r < 8; ++r)
            sacc[mt][r] = 0.f;

    for (int nt = 0; nt < 4; ++nt) {
        const int   n_g = n0 + wn + nt * 16 + nlan;
        const float bvn = bias[n_g];
        for (int mt = 0; mt < 4; ++mt) {
            const v8f cc = c[mt][nt];
            for (int r = 0; r < 8; ++r) {
                const float logit = cc[r] + bvn;
                sacc[mt][r] += __expf(logit);
                if (n_g == tgrow[mt][r]) {   // unique (lane, nt) match in whole grid
                    tlogit[m0 + wm + mt * 16 + r + halfsel * 8] = logit;
                }
            }
        }
    }

    // reduce within each 16-lane half, then one atomic per token row
    for (int mt = 0; mt < 4; ++mt) {
        for (int r = 0; r < 8; ++r) {
            float s = sacc[mt][r];
            for (int off = 1; off < 16; off <<= 1)
                s += __shfl_xor(s, off, 32);
            if (nlan == 0)
                atomicAdd(&sumexp[m0 + wm + mt * 16 + r + halfsel * 8], s);
        }
    }
}

// ---------------------------------------------------------------------------
// Finalize: per-token logp, per-sequence averages, NLL + DPO preference loss
// ---------------------------------------------------------------------------
__global__ __launch_bounds__(256)
void dpo_finalize_kernel(const float* __restrict__ sumexp,
                         const float* __restrict__ tlogit,
                         const int*   __restrict__ tgt,
                         const float* __restrict__ refc,
                         const float* __restrict__ refr,
                         float* __restrict__ out)
{
    __shared__ float reds[256];
    __shared__ float redc[256];
    __shared__ float seq_s[TWO_B];
    __shared__ float seq_c[TWO_B];
    const int tid = threadIdx.x;

    for (int b = 0; b < TWO_B; ++b) {
        float s = 0.f, c = 0.f;
        for (int t = tid; t < SEQ; t += 256) {
            const int m = b * SEQ + t;
            if (tgt[m] != IGNORE_INDEX) {
                s += tlogit[m] - logf(sumexp[m]);
                c += 1.f;
            }
        }
        reds[tid] = s; redc[tid] = c;
        __syncthreads();
        for (int off = 128; off > 0; off >>= 1) {
            if (tid < off) { reds[tid] += reds[tid + off]; redc[tid] += redc[tid + off]; }
            __syncthreads();
        }
        if (tid == 0) { seq_s[b] = reds[0]; seq_c[b] = redc[0]; }
        __syncthreads();
    }

    if (tid == 0) {
        const int half = TWO_B / 2;
        float nll_num = 0.f, nll_den = 0.f, pref = 0.f;
        for (int i = 0; i < half; ++i) { nll_num += seq_s[i]; nll_den += seq_c[i]; }
        const float nll = -nll_num / nll_den;
        for (int i = 0; i < half; ++i) {
            const float ch = seq_s[i] / seq_c[i] - refc[i];
            const float rj = seq_s[half + i] / seq_c[half + i] - refr[i];
            const float x  = BETA * (ch - rj);
            const float ls = fminf(x, 0.f) - log1pf(__expf(-fabsf(x)));  // log_sigmoid
            pref += -ls;
        }
        pref /= (float)half;
        out[0] = pref + nll;
    }
}

// ---------------------------------------------------------------------------
extern "C" void kernel_launch(void* const* d_in, const int* in_sizes, int n_in,
                              void* d_out, int out_size, void* d_ws, size_t ws_size,
                              hipStream_t stream) {
    (void)in_sizes; (void)n_in; (void)out_size; (void)ws_size;
    const float* W    = (const float*)d_in[0];  // lin_weight [VOCAB,HIDDEN]
    const float* X    = (const float*)d_in[1];  // _input     [TWO_B,SEQ,HIDDEN]
    const int*   tgt  = (const int*)  d_in[2];  // target     [TWO_B,SEQ]
    const float* refc = (const float*)d_in[3];  // ref_chosen_logps
    const float* refr = (const float*)d_in[4];  // ref_rejected_logps
    const float* bias = (const float*)d_in[5];  // bias [VOCAB]
    float* out = (float*)d_out;

    // workspace layout: bf16 W (256 MB) | bf16 X (32 MB) | sumexp | tlogit
    unsigned short* Wb = (unsigned short*)d_ws;
    unsigned short* Xb = Wb + (size_t)VOCAB * HIDDEN;
    float* sumexp = (float*)(Xb + (size_t)MTOT * HIDDEN);
    float* tlogit = sumexp + MTOT;

    const int nW8 = (VOCAB * HIDDEN) / 8;   // 16,384,000
    const int nX8 = (MTOT * HIDDEN) / 8;    // 2,097,152
    cvt_bf16_kernel<<<(nW8 + 255) / 256, 256, 0, stream>>>(W, Wb, nW8);
    cvt_bf16_kernel<<<(nX8 + 255) / 256, 256, 0, stream>>>(X, Xb, nX8);

    dpo_zero_kernel<<<(MTOT + 255) / 256, 256, 0, stream>>>(sumexp, tlogit, MTOT);

    dim3 grid(MTOT / MT, VOCAB / NT);       // (16, 125); x-fast => weight tile L2 reuse
    dpo_gemm_kernel<<<grid, 512, 0, stream>>>(Wb, Xb, tgt, bias, sumexp, tlogit);

    dpo_finalize_kernel<<<1, 256, 0, stream>>>(sumexp, tlogit, tgt, refc, refr, out);
}
